// NIFEncoder_18940805775845
// MI455X (gfx1250) — matrix operand order repaired and checked by reading, not compiled, not run on hardware
//
#include <hip/hip_runtime.h>
#include <hip/hip_bf16.h>

typedef __attribute__((ext_vector_type(16))) _Float16 v16h;
typedef __attribute__((ext_vector_type(8)))  float    v8f;
typedef __attribute__((ext_vector_type(4)))  float    f4;

#define NIF_B 4096
#define NIF_L 64
#define NIF_D 128
#define NIF_TROWS 80   // 65 valid count values (0..64) padded to 5 WMMA M-tiles

__device__ __forceinline__ void split_f16(float x, _Float16& hi, _Float16& lo) {
    _Float16 h = (_Float16)x;
    hi = h;
    lo = (_Float16)(x - (float)h);
}

// ---------------------------------------------------------------------------
// Kernel 1: build table[v][n] = relu(v*W1[n'] + b1) @ W2 + b2 for v = 0..79
// via fp32-accurate split-f16 WMMA (hi*hi + hi*lo + lo*hi, f32 accumulate).
// Single workgroup, 8 waves; wave w owns N-tile [16w, 16w+16).
// ---------------------------------------------------------------------------
__global__ __launch_bounds__(256) void nif_table_kernel(
    const float* __restrict__ W1, const float* __restrict__ b1,
    const float* __restrict__ W2, const float* __restrict__ b2,
    float* __restrict__ table) {
    const int lane = threadIdx.x & 31;
    const int wave = threadIdx.x >> 5;   // 0..7
    const int lm   = lane & 15;          // 0..15
    const int half = lane >> 4;          // lane-half selects K sub-range
    const int n    = wave * 16 + lm;     // output column 0..127

    for (int mt = 0; mt < 5; ++mt) {     // M tiles: rows 16*mt .. 16*mt+15
        const float v = (float)(mt * 16 + lm);   // count value for this A row
        v8f c = {};
        for (int kt = 0; kt < 4; ++kt) {         // K steps of 32
            v16h ah, al, bh, bl;
            // A fragment (16x32 f16, ISA layout: lanes 0-15 K=0..7/16..23,
            // lanes 16-31 K=8..15/24..31; 2 halves packed per VGPR)
#pragma unroll
            for (int i = 0; i < 8; ++i) {
                const int k0 = kt * 32 + (i >> 2) * 16 + half * 8 + (i & 3) * 2;
                const float h0 = fmaxf(fmaf(v, W1[k0],     b1[k0]),     0.0f);
                const float h1 = fmaxf(fmaf(v, W1[k0 + 1], b1[k0 + 1]), 0.0f);
                _Float16 hi0, lo0, hi1, lo1;
                split_f16(h0, hi0, lo0);
                split_f16(h1, hi1, lo1);
                ah[2 * i] = hi0; ah[2 * i + 1] = hi1;
                al[2 * i] = lo0; al[2 * i + 1] = lo1;
            }
            // B fragment (32x16 f16: lanes 0-15 hold K=0..7/16..23, lanes 16-31
            // K=8..15/24..31, column n = lane&15, K pairs per VGPR)
#pragma unroll
            for (int j = 0; j < 8; ++j) {
                const int kb = kt * 32 + (j >> 2) * 16 + half * 8 + (j & 3) * 2;
                _Float16 hi0, lo0, hi1, lo1;
                split_f16(W2[kb * NIF_D + n],       hi0, lo0);
                split_f16(W2[(kb + 1) * NIF_D + n], hi1, lo1);
                bh[2 * j] = hi0; bh[2 * j + 1] = hi1;
                bl[2 * j] = lo0; bl[2 * j + 1] = lo1;
            }
            c = __builtin_amdgcn_wmma_f32_16x16x32_f16(false, ah, false, bh,
                                                       (short)0, c, false, false);
            c = __builtin_amdgcn_wmma_f32_16x16x32_f16(false, ah, false, bl,
                                                       (short)0, c, false, false);
            c = __builtin_amdgcn_wmma_f32_16x16x32_f16(false, al, false, bh,
                                                       (short)0, c, false, false);
        }
        const float bias = b2[n];
        // D layout: lanes 0-15 -> M=r, lanes 16-31 -> M=r+8, N=lane&15
#pragma unroll
        for (int r = 0; r < 8; ++r) {
            const int m = mt * 16 + half * 8 + r;
            table[m * NIF_D + n] = c[r] + bias;
        }
    }
}

// ---------------------------------------------------------------------------
// Kernel 2: per batch row — O(L^2) count computation in LDS, then
// out[b,l] = table[count0] + table[count1] gathered from an LDS-staged table.
// Bandwidth-bound on the 256MB output store.
// ---------------------------------------------------------------------------
__global__ __launch_bounds__(256) void nif_main_kernel(
    const int* __restrict__ src_id_g, const int* __restrict__ dst_id_g,
    const int* __restrict__ src_nb_g, const int* __restrict__ dst_nb_g,
    const float* __restrict__ table, float* __restrict__ out) {
    __shared__ __attribute__((aligned(16))) float T[65 * NIF_D];  // 33 KB
    __shared__ int s_nb[NIF_L];
    __shared__ int d_nb[NIF_L];
    __shared__ int app[4][NIF_L];   // [src_c0, src_c1, dst_c0, dst_c1]
    __shared__ int scal[2];         // cd_srcid, cs_dstid

    const int b = blockIdx.x;
    const int t = threadIdx.x;

    // Stage table into LDS (float4)
    for (int i = t; i < (65 * NIF_D) / 4; i += 256)
        ((f4*)T)[i] = ((const f4*)table)[i];
    if (t < NIF_L) {
        s_nb[t] = src_nb_g[b * NIF_L + t];
        d_nb[t] = dst_nb_g[b * NIF_L + t];
    }
    __syncthreads();

    const int sid = src_id_g[b];
    const int did = dst_id_g[b];
    if (t == 0) {
        int cd = 0, cs = 0;
        for (int j = 0; j < NIF_L; ++j) {
            cd += (d_nb[j] == sid);
            cs += (s_nb[j] == did);
        }
        scal[0] = cd;
        scal[1] = cs;
    }
    __syncthreads();

    if (t < NIF_L) {
        const int si = s_nb[t];
        const int di = d_nb[t];
        int c_src = 0, c_dst = 0, c_ds = 0, c_sd = 0;
        for (int j = 0; j < NIF_L; ++j) {
            const int sj = s_nb[j];
            const int dj = d_nb[j];
            c_src += (si == sj);   // count of src nb within src list
            c_dst += (di == dj);   // count of dst nb within dst list
            c_ds  += (si == dj);   // count of src nb within dst list
            c_sd  += (di == sj);   // count of dst nb within src list
        }
        const int  cd_srcid     = scal[0];
        const int  cs_dstid     = scal[1];
        const bool srcid_in_dst = cd_srcid > 0;
        const bool same         = (sid == did);
        const bool cond2        = (cs_dstid > 0) || (same && srcid_in_dst);
        const int  v2           = (same && srcid_in_dst) ? cd_srcid : cs_dstid;
        app[0][t] = c_src;
        app[1][t] = (si == did && cond2) ? v2 : c_ds;
        app[2][t] = (di == sid && srcid_in_dst) ? cd_srcid : c_sd;
        app[3][t] = c_dst;
    }
    __syncthreads();

    // Emit: wave w handles row (8*it + w); 32 lanes cover 128 cols as float4.
    const int row0 = t >> 5;            // 0..7
    const int d4   = (t & 31) * 4;      // 0..124
    const size_t baseS = (size_t)b * (NIF_L * NIF_D);
    const size_t baseD = (size_t)NIF_B * NIF_L * NIF_D + baseS;
#pragma unroll
    for (int it = 0; it < 8; ++it) {
        const int i = it * 8 + row0;
        const f4 s0 = *(const f4*)&T[app[0][i] * NIF_D + d4];
        const f4 s1 = *(const f4*)&T[app[1][i] * NIF_D + d4];
        *(f4*)&out[baseS + i * NIF_D + d4] = s0 + s1;
        const f4 d0 = *(const f4*)&T[app[2][i] * NIF_D + d4];
        const f4 d1 = *(const f4*)&T[app[3][i] * NIF_D + d4];
        *(f4*)&out[baseD + i * NIF_D + d4] = d0 + d1;
    }
}

extern "C" void kernel_launch(void* const* d_in, const int* in_sizes, int n_in,
                              void* d_out, int out_size, void* d_ws, size_t ws_size,
                              hipStream_t stream) {
    const int*   src_id = (const int*)d_in[0];
    const int*   dst_id = (const int*)d_in[1];
    const int*   src_nb = (const int*)d_in[2];
    const int*   dst_nb = (const int*)d_in[3];
    const float* W1     = (const float*)d_in[4];
    const float* b1     = (const float*)d_in[5];
    const float* W2     = (const float*)d_in[6];
    const float* b2     = (const float*)d_in[7];
    float* table = (float*)d_ws;          // 80*128 fp32 = 40 KB scratch
    float* out   = (float*)d_out;

    nif_table_kernel<<<1, 256, 0, stream>>>(W1, b1, W2, b2, table);
    nif_main_kernel<<<NIF_B, 256, 0, stream>>>(src_id, dst_id, src_nb, dst_nb,
                                               table, out);
}
